// GCN_7361573945717
// MI455X (gfx1250) — matrix operand order repaired and checked by reading, not compiled, not run on hardware
//
#include <hip/hip_runtime.h>

// ---------------------------------------------------------------------------
// CDNA5 / gfx1250 GCN-style pipeline:
//   h1 = x @ W1 + b1                 (bf16 WMMA, f32 accum)
//   p1 = KPropagate(h1)              (2-stage masked scatter-add, f32 atomics)
//   h2 = relu(p1) @ W2 + b2          (bf16 WMMA, relu fused into A staging)
//   out = KPropagate(h2)
// ---------------------------------------------------------------------------

typedef __attribute__((ext_vector_type(16))) __bf16 v16bf;
typedef __attribute__((ext_vector_type(8)))  float  v8f;

// Round-half-up f32->bf16, packed pair: lo = even K.
static __device__ __forceinline__ unsigned int pack_bf16(float lo, float hi) {
  unsigned int l = (__float_as_uint(lo) + 0x8000u) >> 16;
  unsigned int h = (__float_as_uint(hi) + 0x8000u) & 0xFFFF0000u;
  return h | l;
}

// Tiled GEMM: C[M,N] = (relu? relu(A) : A)[M,K] * B[K,N] + bias[N]
// 256 threads = 8 waves; block tile 64(M) x 128(N); BK = 32; LDS double-buffered.
// Waves in a 4(M) x 2(N) grid; each wave computes FOUR 16x16 C tiles,
// reusing one A fragment against four B fragments per K step.
// LDS layouts are in *fragment order* so operand loads are contiguous b128:
//   A: row-major pairs,  Alds[row][pair],  stride 20 dwords (80B, 16B-aligned runs)
//   B: col-major pairs,  Blds[col][pair],  stride 20 dwords
#define BM 64
#define BN 128
#define BK 32
#define AST 20
#define BST 20

union Frag { uint4 q[2]; unsigned int u[8]; v16bf v; };

__global__ __launch_bounds__(256) void gcn_gemm_bf16_wmma(
    const float* __restrict__ A, const float* __restrict__ B,
    const float* __restrict__ bias, float* __restrict__ C,
    int M, int N, int K, int reluA)
{
  __shared__ unsigned int Alds[2][BM * AST];   // 2 x 1280 dwords = 10.2 KB
  __shared__ unsigned int Blds[2][BN * BST];   // 2 x 2560 dwords = 20.5 KB

  const int tid   = threadIdx.x;
  const int lane  = tid & 31;          // wave32
  const int wave  = tid >> 5;          // 0..7
  const int mi    = wave & 3;          // 4 wave-rows (M)
  const int ni    = wave >> 2;         // 2 wave-cols, 64 columns each
  const int lrow  = lane & 15;
  const int lhalf = lane >> 4;

  const int blockM = blockIdx.x * BM;
  const int blockN = blockIdx.y * BN;

  v8f acc[4] = {};

  // ---- staging helpers (inlined by the compiler) ----
  auto stage = [&](int k0, int buf) {
    // A tile: 64x32 f32 -> bf16 pairs. 512 float4 quads, 2 per thread.
    #pragma unroll
    for (int i = 0; i < 2; ++i) {
      int q   = i * 256 + tid;
      int r   = q >> 3;                 // row in tile
      int kc4 = (q & 7) * 4;            // K offset (quad)
      int gr  = blockM + r;
      int grc = gr < M - 1 ? gr : M - 1;     // branchless clamp
      float4 v = *(const float4*)(A + (size_t)grc * K + k0 + kc4);
      if (reluA) {                       // uniform branch
        v.x = v.x > 0.0f ? v.x : 0.0f;
        v.y = v.y > 0.0f ? v.y : 0.0f;
        v.z = v.z > 0.0f ? v.z : 0.0f;
        v.w = v.w > 0.0f ? v.w : 0.0f;
      }
      unsigned int d0 = pack_bf16(v.x, v.y);
      unsigned int d1 = pack_bf16(v.z, v.w);
      if (gr >= M) { d0 = 0u; d1 = 0u; }
      uint2 st; st.x = d0; st.y = d1;
      *(uint2*)&Alds[buf][r * AST + (kc4 >> 1)] = st;   // ds_store_b64
    }
    // B tile: 32x128 f32 -> fragment-order bf16 pairs Bt[col][pair].
    // Work item w: pair-row p = w>>5 (K=2p,2p+1), col quad cq = (w&31)*4.
    #pragma unroll
    for (int i = 0; i < 2; ++i) {
      int w  = i * 256 + tid;
      int p  = w >> 5;
      int cq = (w & 31) * 4;
      const float* b0 = B + (size_t)(k0 + 2 * p) * N + blockN + cq;
      float4 ra = *(const float4*)b0;          // row K=2p
      float4 rb = *(const float4*)(b0 + N);    // row K=2p+1
      Blds[buf][(cq + 0) * BST + p] = pack_bf16(ra.x, rb.x);
      Blds[buf][(cq + 1) * BST + p] = pack_bf16(ra.y, rb.y);
      Blds[buf][(cq + 2) * BST + p] = pack_bf16(ra.z, rb.z);
      Blds[buf][(cq + 3) * BST + p] = pack_bf16(ra.w, rb.w);
    }
  };

  const int nk = K / BK;
  stage(0, 0);
  __syncthreads();

  for (int it = 0; it < nk; ++it) {
    const int cur = it & 1;
    if (it + 1 < nk) stage((it + 1) * BK, cur ^ 1);   // prefetch next tile

    // A fragment: two contiguous 16B runs (CDNA5 16-bit A 16x32 striping:
    // v=0..3 -> pair lhalf*4+v ; v=4..7 -> pair 8+lhalf*4+(v-4)).
    Frag af;
    const unsigned int* ab = &Alds[cur][(mi * 16 + lrow) * AST + lhalf * 4];
    af.q[0] = *(const uint4*)(ab);
    af.q[1] = *(const uint4*)(ab + 8);

  // 4 B fragments (each: two 4-dword runs, same striping as A), 4 WMMAs sharing af.
    #pragma unroll
    for (int t = 0; t < 4; ++t) {
      Frag bf;
      const unsigned int* bb =
          &Blds[cur][(ni * 64 + t * 16 + lrow) * BST + lhalf * 4];
      bf.q[0] = *(const uint4*)(bb);
      bf.q[1] = *(const uint4*)(bb + 8);
      acc[t] = __builtin_amdgcn_wmma_f32_16x16x32_bf16(
          false, af.v, false, bf.v, (short)0, acc[t], false, false);
    }
    __syncthreads();   // single barrier per iteration (double buffering)
  }

  // --- store: C/D layout: VGPR i -> rows i (lanes 0-15) / i+8 (lanes 16-31)
  #pragma unroll
  for (int t = 0; t < 4; ++t) {
    int col = blockN + ni * 64 + t * 16 + lrow;
    float bv = bias[col];
    #pragma unroll
    for (int i = 0; i < 8; ++i) {
      int row = blockM + mi * 16 + (lhalf ? (i + 8) : i);
      if (row < M) C[(size_t)row * N + col] = acc[t][i] + bv;
    }
  }
}

// Masked scatter-add: out[dst[e], :] += cur[src[e], :] where mask[e]==target.
// float4 chunks -> coalesced b128 loads + global_atomic_add_f32.
__global__ __launch_bounds__(256) void gcn_scatter_add4(
    const float* __restrict__ cur, float* __restrict__ out,
    const int* __restrict__ src, const int* __restrict__ dst,
    const int* __restrict__ mask, int E, int lgF4, int target)
{
  long long t = (long long)blockIdx.x * 256 + threadIdx.x;
  long long total = (long long)E << lgF4;
  if (t >= total) return;
  int e = (int)(t >> lgF4);
  if (mask[e] != target) return;
  int j = (int)(t & ((1 << lgF4) - 1)) * 4;
  int F = 4 << lgF4;
  const float4 val = *(const float4*)(cur + (size_t)src[e] * F + j);
  float* o = out + (size_t)dst[e] * F + j;
  atomicAdd(o + 0, val.x);
  atomicAdd(o + 1, val.y);
  atomicAdd(o + 2, val.z);
  atomicAdd(o + 3, val.w);
}

extern "C" void kernel_launch(void* const* d_in, const int* in_sizes, int n_in,
                              void* d_out, int out_size, void* d_ws, size_t ws_size,
                              hipStream_t stream) {
  const float* x    = (const float*)d_in[0];
  const int*   ei   = (const int*)d_in[1];
  const int*   mask = (const int*)d_in[2];
  // d_in[3]=vertex_cnt, d_in[4]=rule_cnt (unused; N derived from x)
  const float* W1 = (const float*)d_in[5];
  const float* b1 = (const float*)d_in[6];
  const float* W2 = (const float*)d_in[7];
  const float* b2 = (const float*)d_in[8];

  const int F_IN  = in_sizes[6];            // 512 (len b1)
  const int F_OUT = in_sizes[8];            // 128 (len b2)
  const int Nn    = in_sizes[0] / F_IN;     // 60000
  const int E     = in_sizes[2];            // 160000
  const int* src = ei;
  const int* dst = ei + E;

  char* ws = (char*)d_ws;
  size_t B1 = (size_t)Nn * F_IN  * sizeof(float);   // 122.88 MB
  size_t B2 = (size_t)Nn * F_OUT * sizeof(float);   //  30.72 MB
  float* h1  = (float*)(ws);                // GEMM1 out; later P1 result (OUT)
  float* s01 = (float*)(ws + B1);           // P1 stage-0 aggregate (S0)
  float* h2  = (float*)(ws + 2 * B1);       // GEMM2 out
  float* s02 = (float*)(ws + 2 * B1 + B2);  // P2 stage-0 aggregate
  float* outp = (float*)d_out;

  const int lg1 = __builtin_ctz(F_IN / 4);   // 7
  const int lg2 = __builtin_ctz(F_OUT / 4);  // 5
  dim3 blk(256);

  // h1 = x @ W1 + b1
  dim3 g1((Nn + BM - 1) / BM, F_IN / BN);
  gcn_gemm_bf16_wmma<<<g1, blk, 0, stream>>>(x, W1, b1, h1, Nn, F_IN, F_IN, 0);

  // P1: S0 = scatter_{mask==0}(h1); OUT = S0 + scatter_{mask==1}(S0)
  hipMemsetAsync(s01, 0, B1, stream);
  {
    long long total = (long long)E << lg1;
    int blocks = (int)((total + 255) / 256);
    gcn_scatter_add4<<<blocks, blk, 0, stream>>>(h1, s01, src, dst, mask, E, lg1, 0);
    hipMemcpyAsync(h1, s01, B1, hipMemcpyDeviceToDevice, stream);  // OUT := S0 (h1 dead)
    gcn_scatter_add4<<<blocks, blk, 0, stream>>>(s01, h1, src, dst, mask, E, lg1, 1);
  }

  // h2 = relu(P1) @ W2 + b2   (relu fused into A staging)
  dim3 g2((Nn + BM - 1) / BM, F_OUT / BN);
  gcn_gemm_bf16_wmma<<<g2, blk, 0, stream>>>(h1, W2, b2, h2, Nn, F_OUT, F_IN, 1);

  // P2 into d_out
  hipMemsetAsync(s02, 0, B2, stream);
  {
    long long total = (long long)E << lg2;
    int blocks = (int)((total + 255) / 256);
    gcn_scatter_add4<<<blocks, blk, 0, stream>>>(h2, s02, src, dst, mask, E, lg2, 0);
    hipMemcpyAsync(outp, s02, B2, hipMemcpyDeviceToDevice, stream);
    gcn_scatter_add4<<<blocks, blk, 0, stream>>>(s02, outp, src, dst, mask, E, lg2, 1);
  }
}
